// NonLocalSparseAttention_1374389535064
// MI455X (gfx1250) — matrix operand order repaired and checked by reading, not compiled, not run on hardware
//
#include <hip/hip_runtime.h>
#include <hip/hip_bf16.h>
#include <math.h>

// ---------------------------------------------------------------------------
// NonLocalSparseAttention for MI455X (gfx1250, wave32, WMMA + async-to-LDS).
//
// Shapes (fixed by reference): N=4, Cin=64, H=W=96 -> L=9216, C=16,
// N_HASHES=4, hash_buckets=64, CHUNK=144, chunks/hash=64, padding=0.
//
// Roofline: attention core ~10.2 GFLOP (90% of FLOPs), traffic <100 MB
// (~4us @ 23.3 TB/s) -> launch/bandwidth bound. Dominant matmuls on
// v_wmma_f32_16x16x32_f16 (channel dim 16 zero-padded to K=32), flash-style
// online softmax keeps each 432-key bucket LDS-resident (54 KB/WG).
// y_embed is produced in f16 by the WMMA 1x1-conv so the attention V staging
// is a raw byte copy -> GLOBAL_LOAD_ASYNC_TO_LDS_B128 (ASYNCcnt path),
// drained with s_wait_asynccnt. Counting sort (codes in [0,256)) reproduces
// stable jnp.argsort; attention output is scattered straight to the unsorted
// position (index % L), removing undo_sort.
// ---------------------------------------------------------------------------

typedef __attribute__((ext_vector_type(16))) _Float16 v16h;
typedef __attribute__((ext_vector_type(8)))  float    v8f;

#define NB    4       // batch
#define CIN   64
#define HW    96
#define LPIX  9216    // 96*96
#define CXE   16      // x_embed channels
#define NH    4       // hashes
#define NBKT  64      // hash buckets / chunks per hash
#define CHK   144     // chunk size (queries per bucket)
#define KEYS  432     // 3*CHK keys per bucket
#define KEYP  448     // padded to 14 tiles of 32
#define TOTJ  (NH*LPIX)   // 36864 elements sorted per batch

static __device__ inline v8f wmma16(v16h a, v16h b, v8f c) {
  return __builtin_amdgcn_wmma_f32_16x16x32_f16(
      /*neg_a=*/false, a, /*neg_b=*/false, b,
      /*c_mod=*/(short)0, c, /*reuse_a=*/false, /*reuse_b=*/false);
}

// --------------------------- conv 3x3, 64->16 -------------------------------
__global__ __launch_bounds__(256) void conv3x3_kernel(
    const float* __restrict__ in, const float* __restrict__ w,
    const float* __restrict__ bias, float* __restrict__ xe) {
  int g = blockIdx.x * blockDim.x + threadIdx.x;
  if (g >= NB * LPIX) return;
  int b = g / LPIX, p = g % LPIX;
  int y = p / HW, x = p % HW;
  float acc[CXE];
  #pragma unroll
  for (int c = 0; c < CXE; ++c) acc[c] = bias[c];
  for (int ci = 0; ci < CIN; ++ci) {
    const float* ip = in + ((size_t)b * CIN + ci) * LPIX;
    float pix[9];
    int t = 0;
    #pragma unroll
    for (int dy = -1; dy <= 1; ++dy)
      #pragma unroll
      for (int dx = -1; dx <= 1; ++dx) {
        int yy = y + dy, xx = x + dx;
        pix[t++] = (yy >= 0 && yy < HW && xx >= 0 && xx < HW) ? ip[yy * HW + xx] : 0.f;
      }
    #pragma unroll
    for (int co = 0; co < CXE; ++co) {
      const float* wp = w + ((size_t)co * CIN + ci) * 9;
      float a = 0.f;
      #pragma unroll
      for (int k = 0; k < 9; ++k) a += pix[k] * wp[k];
      acc[co] += a;
    }
  }
  float* op = xe + ((size_t)b * LPIX + p) * CXE;
  #pragma unroll
  for (int c = 0; c < CXE; ++c) op[c] = acc[c];
}

// ------------------- conv 1x1, 64->64 as WMMA GEMM (f16 out) ----------------
// y[b][p][co] = bias[co] + sum_ci in[b][ci][p] * w[co][ci], stored as f16 so
// the attention V staging is a raw byte copy (async-to-LDS eligible).
__global__ __launch_bounds__(128) void conv1x1_kernel(
    const float* __restrict__ in, const float* __restrict__ w,
    const float* __restrict__ bias, _Float16* __restrict__ ye) {
  int gw   = blockIdx.x * 4 + (threadIdx.x >> 5);   // 0..2303
  int lane = threadIdx.x & 31;
  int b  = gw / (LPIX / 16);
  int mt = gw % (LPIX / 16);
  int p0 = mt * 16;
  int prow = p0 + (lane & 15);

  // A-matrix (16x32 f16): lanes<16 hold K {0..15}; lanes>=16 {16..31}
  v16h a0, a1;
  int akb = (lane >= 16) ? 16 : 0;
  #pragma unroll
  for (int j = 0; j < 16; ++j) {
    int r  = j >> 1;
    int kc = akb + j;  // 0..31
    a0[j] = (_Float16)in[((size_t)b * CIN + kc) * LPIX + prow];
    a1[j] = (_Float16)in[((size_t)b * CIN + 32 + kc) * LPIX + prow];
  }
  int bkb = (lane >= 16) ? 16 : 0;
  int rowoff = (lane >= 16) ? 8 : 0;
  #pragma unroll
  for (int t = 0; t < 4; ++t) {
    int co = t * 16 + (lane & 15);
    v8f o;
    #pragma unroll
    for (int v = 0; v < 8; ++v) o[v] = bias[co];
    // B-matrix (32x16 f16): VGPR r holds K {2r,2r+1} (+16 for upper lanes), N=lane&15
    v16h b0, b1;
    #pragma unroll
    for (int j = 0; j < 16; ++j) {
      int ci = bkb + 2 * (j >> 1) + (j & 1);
      b0[j] = (_Float16)w[(size_t)co * CIN + ci];
      b1[j] = (_Float16)w[(size_t)co * CIN + 32 + ci];
    }
    o = wmma16(a0, b0, o);
    o = wmma16(a1, b1, o);
    #pragma unroll
    for (int v = 0; v < 8; ++v)
      ye[((size_t)b * LPIX + p0 + v + rowoff) * CIN + co] = (_Float16)o[v];
  }
}

// ----------------------------- LSH hashing ----------------------------------
__global__ __launch_bounds__(256) void hash_kernel(
    const float* __restrict__ xe, const float* __restrict__ rot,
    int* __restrict__ hc) {
  int g = blockIdx.x * blockDim.x + threadIdx.x;
  if (g >= NB * NH * LPIX) return;
  int b = g / (NH * LPIX);
  int r = g % (NH * LPIX);
  int h = r / LPIX;
  int l = r % LPIX;
  const float* xr = xe + ((size_t)b * LPIX + l) * CXE;
  float rv[32];
  #pragma unroll
  for (int i = 0; i < 32; ++i) rv[i] = 0.f;
  for (int c = 0; c < CXE; ++c) {
    float xv = xr[c];
    const float* rp = rot + ((size_t)c * NH + h) * 32;
    #pragma unroll
    for (int i = 0; i < 32; ++i) rv[i] += xv * rp[i];
  }
  // argmax over [rv, -rv], first occurrence wins (strict >)
  float best = -3.0e38f; int bi = 0;
  #pragma unroll
  for (int i = 0; i < 32; ++i) { if (rv[i] > best) { best = rv[i]; bi = i; } }
  #pragma unroll
  for (int i = 0; i < 32; ++i) { float v = -rv[i]; if (v > best) { best = v; bi = 32 + i; } }
  hc[(size_t)b * TOTJ + h * LPIX + l] = bi + h * NBKT;
}

// -------------------- stable counting sort (codes in [0,256)) ---------------
__global__ __launch_bounds__(256) void zero_kernel(int* p, int n) {
  int g = blockIdx.x * blockDim.x + threadIdx.x;
  if (g < n) p[g] = 0;
}

__global__ __launch_bounds__(256) void sort_build(
    const int* __restrict__ hc, int* __restrict__ hist) {
  int b = blockIdx.x, t = threadIdx.x;                 // grid=4, block=256
  int* hp = hist + ((size_t)b * 256 + t) * 256;        // private region
  const int* cp = hc + (size_t)b * TOTJ + t * (TOTJ / 256);
  for (int e = 0; e < TOTJ / 256; ++e) ++hp[cp[e]];
}

__global__ __launch_bounds__(256) void sort_scan(int* __restrict__ hist) {
  __shared__ int tot[256];
  __shared__ int base[256];
  int b = blockIdx.x, c = threadIdx.x;
  int* hp = hist + (size_t)b * 256 * 256;
  int s = 0;
  for (int t = 0; t < 256; ++t) s += hp[t * 256 + c];
  tot[c] = s;
  __syncthreads();
  if (c == 0) {
    int run = 0;
    for (int i = 0; i < 256; ++i) { base[i] = run; run += tot[i]; }
  }
  __syncthreads();
  int run = base[c];
  for (int t = 0; t < 256; ++t) { int v = hp[t * 256 + c]; hp[t * 256 + c] = run; run += v; }
}

__global__ __launch_bounds__(256) void sort_scatter(
    const int* __restrict__ hc, int* __restrict__ hist, int* __restrict__ srt) {
  int b = blockIdx.x, t = threadIdx.x;
  int* hp = hist + ((size_t)b * 256 + t) * 256;
  const int chunk = TOTJ / 256;
  const int* cp = hc + (size_t)b * TOTJ + t * chunk;
  int* sp = srt + (size_t)b * TOTJ;
  int jb = t * chunk;
  for (int e = 0; e < chunk; ++e) {
    int pos = hp[cp[e]]++;
    sp[pos] = jb + e;                 // stable: chunks are in element order
  }
}

// --------------------- bucket attention (flash-style, WMMA) -----------------
// One WG of 9 waves per (b, hash, bucket). Wave w owns query rows 16w..16w+15.
// Q = raw x rows (f16, channel dim padded 16->32), K = row-normalized x rows,
// V = y rows (f16, async-copied to LDS). Online softmax over 14 key-tiles of
// 32; exact LSE saved. Output scattered directly to UNSORTED position.
__global__ __launch_bounds__(288) void attn_kernel(
    const float* __restrict__ xe, const _Float16* __restrict__ ye,
    const int* __restrict__ srt, float* __restrict__ ret,
    float* __restrict__ lse_out) {
  __shared__ int      s_idx[KEYP];          // original pixel per key
  __shared__ _Float16 s_k[KEYP][CXE];       // normalized keys (f16)
  __shared__ _Float16 s_v[224][CIN];        // V staged in 2 halves of 224 keys
  __shared__ _Float16 s_p[9][16][32];       // per-wave exp(S) bounce buffer

  int wg   = blockIdx.x;          // 0..1023
  int kb_  = wg & 63;             // bucket
  int h    = (wg >> 6) & 3;
  int b    = wg >> 8;
  int tid  = threadIdx.x;
  int lane = tid & 31;
  int wave = tid >> 5;

  const size_t base = (size_t)b * TOTJ + (size_t)h * LPIX;
  // keys 0..143 = bucket k, 144..287 = k-1 (wrap), 288..431 = k+1 (wrap)
  for (int i = tid; i < KEYP; i += 288) {
    int kk, r;
    if (i < 144)      { kk = kb_;              r = i; }
    else if (i < 288) { kk = (kb_ + 63) & 63;  r = i - 144; }
    else if (i < 432) { kk = (kb_ + 1) & 63;   r = i - 288; }
    else              { kk = kb_;              r = 0; }     // pad (masked)
    int oi = srt[base + (size_t)kk * CHK + r] % LPIX;
    s_idx[i] = oi;
    // warm the scattered x_embed rows (global_prefetch_b8)
    __builtin_prefetch(xe + ((size_t)b * LPIX + oi) * CXE, 0, 0);
  }
  __syncthreads();

  // normalized keys -> f16
  for (int i = tid; i < KEYP; i += 288) {
    const float* row = xe + ((size_t)b * LPIX + s_idx[i]) * CXE;
    float vv[CXE]; float ss = 0.f;
    #pragma unroll
    for (int c = 0; c < CXE; ++c) { vv[c] = row[c]; ss += vv[c] * vv[c]; }
    float inv = 1.0f / fmaxf(sqrtf(ss), 5e-5f);
    bool valid = (i < KEYS);
    #pragma unroll
    for (int c = 0; c < CXE; ++c)
      s_k[i][c] = valid ? (_Float16)(vv[c] * inv) : (_Float16)0.f;
  }
  __syncthreads();

  // Q A-tile (loop-invariant): row M = lane&15, channel k = 16*(lane/16) + j (contiguous order)
  v16h a_q;
  {
    int q = wave * 16 + (lane & 15);
    const float* row = xe + ((size_t)b * LPIX + s_idx[q]) * CXE;
    int akb = (lane >= 16) ? 16 : 0;
    #pragma unroll
    for (int j = 0; j < 16; ++j) {
      int r  = j >> 1;
      int kc = akb + j;
      a_q[j] = (kc < CXE) ? (_Float16)row[kc] : (_Float16)0.f;
    }
  }

  float mrow[8], lsum[8];
  #pragma unroll
  for (int v = 0; v < 8; ++v) { mrow[v] = -3.0e38f; lsum[v] = 0.f; }
  v8f o0 = {}, o1 = {}, o2 = {}, o3 = {};

  int grp  = (lane >= 16) ? 1 : 0;
  int nidx = lane & 15;

  for (int half = 0; half < 2; ++half) {
    // stage this half's V rows: raw f16 copy -> GLOBAL_LOAD_ASYNC_TO_LDS_B128
    // (ASYNCcnt path, no VGPR staging); padded keys zero-filled via DS store.
    for (int e = tid; e < 224 * 8; e += 288) {   // 8 x 16B chunks per 128B row
      int i = e >> 3, part = e & 7;
      int gk = half * 224 + i;
      unsigned lds = (unsigned)(unsigned long long)(void*)&s_v[i][part * 8];
      if (gk < KEYS) {
        const _Float16* src = ye + ((size_t)b * LPIX + s_idx[gk]) * CIN + part * 8;
        asm volatile("global_load_async_to_lds_b128 %0, %1, off"
                     :: "v"(lds), "v"((unsigned long long)(const void*)src)
                     : "memory");
      } else {
        int4 z; z.x = 0; z.y = 0; z.z = 0; z.w = 0;
        *reinterpret_cast<int4*>(&s_v[i][part * 8]) = z;
      }
    }
    asm volatile("s_wait_asynccnt 0" ::: "memory");  // drain async copies
    __syncthreads();

    for (int jt = 0; jt < 7; ++jt) {
      int gkb = half * 224 + jt * 32;   // global key base of this 32-tile
      int vkb = jt * 32;                // base within s_v

      // B-tiles for S: channels x keys, channels>=16 are zero padding
      v16h bk0, bk1;
      {
        int cb = grp ? 16 : 0;
        #pragma unroll
        for (int j = 0; j < 16; ++j) {
          int kc = cb + 2 * (j >> 1) + (j & 1);
          bk0[j] = (kc < CXE) ? s_k[gkb + nidx][kc]      : (_Float16)0.f;
          bk1[j] = (kc < CXE) ? s_k[gkb + 16 + nidx][kc] : (_Float16)0.f;
        }
      }
      v8f zc = {};
      v8f s0 = wmma16(a_q, bk0, zc);
      v8f s1 = wmma16(a_q, bk1, zc);

      // mask padded keys (only last tile has any)
      bool ok0 = (gkb + nidx) < KEYS;
      bool ok1 = (gkb + 16 + nidx) < KEYS;
      float scale[8];
      #pragma unroll
      for (int v = 0; v < 8; ++v) {
        float x0 = ok0 ? s0[v] : -3.0e38f;
        float x1 = ok1 ? s1[v] : -3.0e38f;
        float t = fmaxf(x0, x1);
        t = fmaxf(t, __shfl_xor(t, 1));
        t = fmaxf(t, __shfl_xor(t, 2));
        t = fmaxf(t, __shfl_xor(t, 4));
        t = fmaxf(t, __shfl_xor(t, 8));
        float nm = fmaxf(mrow[v], t);
        scale[v] = __expf(mrow[v] - nm);
        float e0 = ok0 ? __expf(x0 - nm) : 0.f;
        float e1 = ok1 ? __expf(x1 - nm) : 0.f;
        float rs = e0 + e1;
        rs += __shfl_xor(rs, 1);
        rs += __shfl_xor(rs, 2);
        rs += __shfl_xor(rs, 4);
        rs += __shfl_xor(rs, 8);
        lsum[v] = lsum[v] * scale[v] + rs;
        mrow[v] = nm;
        int row = v + grp * 8;
        s_p[wave][row][nidx]      = (_Float16)e0;   // D-layout -> row-major LDS
        s_p[wave][row][16 + nidx] = (_Float16)e1;
      }
      __syncthreads();   // intra-wave LDS RAW ordering (uniform across block)

      // P A-tile from bounce buffer (A-layout remap)
      v16h a_p;
      {
        int row = lane & 15;
        int akb = grp ? 16 : 0;
        #pragma unroll
        for (int j = 0; j < 16; ++j) {
          int r  = j >> 1;
          int kk = akb + j;
          a_p[j] = s_p[wave][row][kk];
        }
      }
      #pragma unroll
      for (int v = 0; v < 8; ++v) {
        o0[v] *= scale[v]; o1[v] *= scale[v];
        o2[v] *= scale[v]; o3[v] *= scale[v];
      }
      // P @ V: 4 channel tiles
      int cb = grp ? 16 : 0;
      #pragma unroll
      for (int t = 0; t < 4; ++t) {
        v16h bv;
        int ch = t * 16 + nidx;
        #pragma unroll
        for (int j = 0; j < 16; ++j) {
          int kk = cb + 2 * (j >> 1) + (j & 1);
          bv[j] = s_v[vkb + kk][ch];
        }
        if (t == 0)      o0 = wmma16(a_p, bv, o0);
        else if (t == 1) o1 = wmma16(a_p, bv, o1);
        else if (t == 2) o2 = wmma16(a_p, bv, o2);
        else             o3 = wmma16(a_p, bv, o3);
      }
      __syncthreads();   // protect next iteration's s_p / s_v writes
    }
  }

  // epilogue: normalize, scatter to unsorted position (index % L)
  #pragma unroll
  for (int v = 0; v < 8; ++v) {
    float inv = 1.0f / lsum[v];
    int row = v + grp * 8;
    int q   = wave * 16 + row;
    int l   = s_idx[q];
    size_t ro = ((((size_t)b * NH + h) * LPIX) + l) * CIN;
    ret[ro + 0 * 16 + nidx] = o0[v] * inv;
    ret[ro + 1 * 16 + nidx] = o1[v] * inv;
    ret[ro + 2 * 16 + nidx] = o2[v] * inv;
    ret[ro + 3 * 16 + nidx] = o3[v] * inv;
    if (nidx == 0)
      lse_out[((size_t)b * NH + h) * LPIX + l] = mrow[v] + __logf(lsum[v]);
  }
}

// -------------------- cross-hash softmax + residual -------------------------
__global__ __launch_bounds__(128) void combine_kernel(
    const float* __restrict__ in, const float* __restrict__ ret,
    const float* __restrict__ lse, float* __restrict__ out) {
  int g = blockIdx.x * blockDim.x + threadIdx.x;
  if (g >= NB * LPIX) return;
  int b = g / LPIX, l = g % LPIX;
  float s[NH], mx = -3.0e38f;
  #pragma unroll
  for (int h = 0; h < NH; ++h) {
    s[h] = lse[((size_t)b * NH + h) * LPIX + l];
    mx = fmaxf(mx, s[h]);
  }
  float den = 0.f;
  #pragma unroll
  for (int h = 0; h < NH; ++h) { s[h] = __expf(s[h] - mx); den += s[h]; }
  float invd = 1.f / den;
  for (int c = 0; c < CIN; ++c) {
    float acc = 0.f;
    #pragma unroll
    for (int h = 0; h < NH; ++h)
      acc += ret[((((size_t)b * NH + h) * LPIX) + l) * CIN + c] * s[h];
    out[((size_t)b * CIN + c) * LPIX + l] =
        in[((size_t)b * CIN + c) * LPIX + l] + acc * invd;
  }
}

// ----------------------------- launcher -------------------------------------
// workspace layout (bytes):
static const size_t OFF_XE   = 0;                                  //  2,359,296
static const size_t OFF_YE   = OFF_XE  + (size_t)NB*LPIX*CXE*4;    //  4,718,592 (f16)
static const size_t OFF_HC   = OFF_YE  + (size_t)NB*LPIX*CIN*2;    //    589,824
static const size_t OFF_SRT  = OFF_HC  + (size_t)NB*TOTJ*4;        //    589,824
static const size_t OFF_HIST = OFF_SRT + (size_t)NB*TOTJ*4;        //  1,048,576
static const size_t OFF_RET  = OFF_HIST+ (size_t)NB*256*256*4;     // 37,748,736
static const size_t OFF_LSE  = OFF_RET + (size_t)NB*NH*LPIX*CIN*4; //    589,824
// total ~47.6 MB

extern "C" void kernel_launch(void* const* d_in, const int* in_sizes, int n_in,
                              void* d_out, int out_size, void* d_ws, size_t ws_size,
                              hipStream_t stream) {
  (void)in_sizes; (void)n_in; (void)out_size; (void)ws_size;
  const float* input   = (const float*)d_in[0];
  const float* w_match = (const float*)d_in[1];
  const float* b_match = (const float*)d_in[2];
  const float* w_asm   = (const float*)d_in[3];
  const float* b_asm   = (const float*)d_in[4];
  const float* rot     = (const float*)d_in[5];
  float* out = (float*)d_out;

  char* ws = (char*)d_ws;
  float*     xe   = (float*)    (ws + OFF_XE);
  _Float16*  ye   = (_Float16*) (ws + OFF_YE);
  int*       hc   = (int*)      (ws + OFF_HC);
  int*       srt  = (int*)      (ws + OFF_SRT);
  int*       hist = (int*)      (ws + OFF_HIST);
  float*     ret  = (float*)    (ws + OFF_RET);
  float*     lse  = (float*)    (ws + OFF_LSE);

  conv3x3_kernel<<<(NB * LPIX + 255) / 256, 256, 0, stream>>>(input, w_match, b_match, xe);
  conv1x1_kernel<<<(NB * LPIX / 16) / 4, 128, 0, stream>>>(input, w_asm, b_asm, ye);
  hash_kernel<<<(NB * NH * LPIX + 255) / 256, 256, 0, stream>>>(xe, rot, hc);
  zero_kernel<<<(NB * 256 * 256 + 255) / 256, 256, 0, stream>>>(hist, NB * 256 * 256);
  sort_build<<<NB, 256, 0, stream>>>(hc, hist);
  sort_scan<<<NB, 256, 0, stream>>>(hist);
  sort_scatter<<<NB, 256, 0, stream>>>(hc, hist, srt);
  attn_kernel<<<NB * NH * NBKT, 288, 0, stream>>>(xe, ye, srt, ret, lse);
  combine_kernel<<<(NB * LPIX + 127) / 128, 128, 0, stream>>>(input, ret, lse, out);
}
